// SelfAttention_34291018891754
// MI455X (gfx1250) — matrix operand order repaired and checked by reading, not compiled, not run on hardware
//
#include <hip/hip_runtime.h>
#include <hip/hip_bf16.h>

// ---------------------------------------------------------------------------
// Self-attention block (GroupNorm -> QKV -> MHA -> proj -> residual)
// N=16, C=512, H=W=32 (S=1024), heads=8, head_dim=64, groups=32.
// GEMMs: v_wmma_f32_16x16x32_bf16 (wave32).
// Shared operands staged with global_load_async_to_lds_b128 (ASYNCcnt),
// double-buffered against the WMMA pipeline.
// ---------------------------------------------------------------------------

typedef __attribute__((ext_vector_type(16))) __bf16 v16bf;
typedef __attribute__((ext_vector_type(8)))  float  v8f;

#define WMMA_BF16(a, b, c) \
    __builtin_amdgcn_wmma_f32_16x16x32_bf16(false, (a), false, (b), (short)0, (c), false, false)

static constexpr int kN = 16;
static constexpr int kC = 512;
static constexpr int kS = 1024;   // H*W
static constexpr int kHeads = 8;
static constexpr int kHD = 64;
static constexpr int kGroups = 32;
static constexpr int kCPG = kC / kGroups; // 16
static constexpr int kKC32 = kC / 32;     // 16 K-chunks per GEMM

// Async 16B copy global -> LDS (tracked by ASYNCcnt). Low 32 bits of a generic
// LDS pointer are the LDS byte address (aperture check discards the rest).
__device__ __forceinline__ void async_copy16(void* lds_dst, const void* src) {
  unsigned l = (unsigned)(unsigned long long)lds_dst;
  asm volatile("global_load_async_to_lds_b128 %0, %1, off" ::"v"(l), "v"(src)
               : "memory");
}
__device__ __forceinline__ void wait_async0() {
  asm volatile("s_wait_asynccnt 0x0" ::: "memory");
}
__device__ __forceinline__ void wait_ds0() {
  asm volatile("s_wait_dscnt 0x0" ::: "memory");
}

// ---------------------------------------------------------------------------
// GroupNorm pass 1: per (n, g) mean / rstd over 16 channels x 1024 pixels.
// ---------------------------------------------------------------------------
__global__ void gn_stats(const float* __restrict__ x, float* __restrict__ mean,
                         float* __restrict__ rstd) {
  const int n = blockIdx.x >> 5;
  const int g = blockIdx.x & 31;
  const float* base = x + ((size_t)n * kC + (size_t)g * kCPG) * kS;
  const int total = kCPG * kS; // 16384
  float s = 0.f, s2 = 0.f;
  for (int i = threadIdx.x; i < total; i += blockDim.x) {
    float v = base[i];
    s += v;
    s2 += v * v;
  }
  #pragma unroll
  for (int m = 1; m < 32; m <<= 1) {
    s  += __shfl_xor(s, m, 32);
    s2 += __shfl_xor(s2, m, 32);
  }
  __shared__ float red[2][8];
  const int w = threadIdx.x >> 5;
  if ((threadIdx.x & 31) == 0) { red[0][w] = s; red[1][w] = s2; }
  __syncthreads();
  if (threadIdx.x == 0) {
    float ts = 0.f, t2 = 0.f;
    #pragma unroll
    for (int i = 0; i < 8; ++i) { ts += red[0][i]; t2 += red[1][i]; }
    const float inv = 1.0f / (float)total;
    const float m = ts * inv;
    const float var = t2 * inv - m * m;
    mean[blockIdx.x] = m;
    rstd[blockIdx.x] = rsqrtf(var + 1e-5f);
  }
}

// ---------------------------------------------------------------------------
// GroupNorm pass 2: h_bf16[n][s][c] = (x[n][c][s]-mean)*rstd*w[c]+b[c]
// ---------------------------------------------------------------------------
__global__ void gn_apply(const float* __restrict__ x, const float* __restrict__ mean,
                         const float* __restrict__ rstd, const float* __restrict__ w,
                         const float* __restrict__ b, __bf16* __restrict__ h) {
  const int n = blockIdx.x >> 9;
  const int c = blockIdx.x & (kC - 1);
  const int g = c >> 4;
  const float m = mean[n * kGroups + g];
  const float r = rstd[n * kGroups + g];
  const float sc = w[c] * r;
  const float bb = b[c] - m * sc;
  const float* xr = x + ((size_t)n * kC + c) * kS;
  __bf16* hb = h + (size_t)n * kS * kC + c;
  for (int s = threadIdx.x; s < kS; s += blockDim.x)
    hb[(size_t)s * kC] = (__bf16)(xr[s] * sc + bb);
}

// ---------------------------------------------------------------------------
// Pack W [O][K] (f32) into per-(strip,kc) contiguous WMMA-B panels (bf16):
//   P[strip][kc][j<4][lane<32][e<16],  o = strip*64 + j*16 + (lane&15),
//                                      k = kc*32 + 16*(lane>>4) + e
// Each (strip,kc) panel is 4KB and is async-staged to LDS whole.
// ---------------------------------------------------------------------------
__global__ void pack_b_panels(const float* __restrict__ W, __bf16* __restrict__ P,
                              int O, int K) {
  const int idx = blockIdx.x * blockDim.x + threadIdx.x;
  if (idx >= O * K) return;
  const int e     = idx & 15;
  const int lane  = (idx >> 4) & 31;
  const int j     = (idx >> 9) & 3;
  const int kc    = (idx >> 11) & (kKC32 - 1);
  const int strip = idx >> 15;
  const int o = strip * 64 + j * 16 + (lane & 15);
  const int k = kc * 32 + ((lane >> 4) << 4) + e;
  P[idx] = (__bf16)W[(size_t)o * K + k];
}

// ---------------------------------------------------------------------------
// QKV GEMM: one block = one o-strip(64) x eight 16-row s-tiles (one per wave).
// Weight panels double-buffered in LDS via async copies (all 8 waves share).
// Scatters into q:[nh][S][64], k:[nh][S][64], v:[nh][64][S].
// ---------------------------------------------------------------------------
__global__ void qkv_gemm(const __bf16* __restrict__ h, const __bf16* __restrict__ Pw,
                         const float* __restrict__ bias, __bf16* __restrict__ q,
                         __bf16* __restrict__ k, __bf16* __restrict__ v) {
  __shared__ __bf16 bstage[2][2048]; // 2 x 4KB panels
  const int wid  = threadIdx.x >> 5;
  const int lane = threadIdx.x & 31;
  const int oStrip = blockIdx.x % 24;
  const int sGroup = blockIdx.x / 24;        // 0..127
  const int sTileG = sGroup * 8 + wid;       // 0..1023
  const int n  = sTileG >> 6;
  const int s0 = (sTileG & 63) * 16;
  const int hl = lane >> 4;
  const int col = lane & 15;
  const int koff = 8 * hl;

  const __bf16* pbase = Pw + (size_t)oStrip * kKC32 * 2048;
  // This wave's 512B slice of each 4KB panel.
  const int sliceOff = wid * 256 + lane * 8; // elements
  async_copy16(&bstage[0][sliceOff], pbase + sliceOff);

  v8f acc[4] = {v8f{}, v8f{}, v8f{}, v8f{}};
  const __bf16* arow = h + ((size_t)n * kS + s0 + col) * kC;

  for (int kc = 0; kc < kKC32; ++kc) {
    wait_async0();
    __syncthreads();
    if (kc + 1 < kKC32)
      async_copy16(&bstage[(kc + 1) & 1][sliceOff],
                   pbase + (size_t)(kc + 1) * 2048 + sliceOff);
    __builtin_prefetch(arow + (kc + 2) * 32, 0, 1);
    v16bf a;
    const __bf16* ap = arow + kc * 32;
    #pragma unroll
    for (int i = 0; i < 8; ++i) {
      a[i]     = ap[2 * koff + i];
      a[8 + i] = ap[2 * koff + 8 + i];
    }
    const __bf16* pan = &bstage[kc & 1][0];
    #pragma unroll
    for (int j = 0; j < 4; ++j) {
      const v16bf* bp = (const v16bf*)(pan + j * 512 + lane * 16);
      acc[j] = WMMA_BF16(a, *bp, acc[j]);
    }
  }

  #pragma unroll
  for (int j = 0; j < 4; ++j) {
    const int o = oStrip * 64 + j * 16 + col;
    const float bo = bias[o];
    #pragma unroll
    for (int r = 0; r < 8; ++r) {
      const float val = acc[j][r] + bo;
      const int s = s0 + r + 8 * hl;
      if (o < kC) {
        const int hh = o >> 6, d = o & 63;
        q[(((size_t)n * kHeads + hh) * kS + s) * kHD + d] = (__bf16)val;
      } else if (o < 2 * kC) {
        const int oo = o - kC;
        const int hh = oo >> 6, d = oo & 63;
        k[(((size_t)n * kHeads + hh) * kS + s) * kHD + d] = (__bf16)val;
      } else {
        const int oo = o - 2 * kC;
        const int hh = oo >> 6, d = oo & 63;
        v[(((size_t)n * kHeads + hh) * kHD + d) * kS + s] = (__bf16)val;
      }
    }
  }
}

// ---------------------------------------------------------------------------
// Flash attention: block = one (n,h), 4 waves x 16 query rows.
// Per 32-wide t-step: K tile (32x64, 4KB) + V tile (64x32, 4KB) async-staged
// to LDS (shared by the 4 waves), double-buffered. Online softmax with
// shfl_xor row reductions; P transposed D->A layout through LDS.
// ---------------------------------------------------------------------------
__global__ void attn_kernel(const __bf16* __restrict__ q, const __bf16* __restrict__ kk,
                            const __bf16* __restrict__ vv, __bf16* __restrict__ ao) {
  __shared__ __bf16 kbuf[2][32 * 64]; // [t_local][d]
  __shared__ __bf16 vbuf[2][64 * 32]; // [d][t_local]
  __shared__ __bf16 pbuf[4][16][32];
  const int wid  = threadIdx.x >> 5;
  const int lane = threadIdx.x & 31;
  const int nh = blockIdx.x >> 4;          // 0..127
  const int s0 = ((blockIdx.x & 15) * 4 + wid) * 16;
  const int hl = lane >> 4;
  const int col = lane & 15;
  const int koff = 8 * hl;

  const __bf16* qb = q  + (size_t)nh * kS * kHD;
  const __bf16* kb = kk + (size_t)nh * kS * kHD;
  const __bf16* vb = vv + (size_t)nh * kHD * kS;

  // Stage one 32-t chunk of K and V into buffer `buf` (4 async b128 per wave).
  auto issue = [&](int t0, int buf) {
    #pragma unroll
    for (int ii = 0; ii < 2; ++ii) {
      const int i = wid + 4 * ii;
      {
        // K: rows t_local = 4i..4i+3 (128B each), 8 lanes per row.
        const int row = 4 * i + (lane >> 3);
        const int part = (lane & 7) * 8;
        async_copy16(&kbuf[buf][row * 64 + part],
                     kb + (size_t)(t0 + row) * kHD + part);
      }
      {
        // V: rows d = 8i..8i+7 (64B each), 4 lanes per row.
        const int row = 8 * i + (lane >> 2);
        const int part = (lane & 3) * 8;
        async_copy16(&vbuf[buf][row * 32 + part],
                     vb + (size_t)row * kS + t0 + part);
      }
    }
  };

  // Q strip (16 x 64) resident for the whole pass.
  v16bf aq[2];
  {
    const __bf16* qr = qb + (size_t)(s0 + col) * kHD;
    #pragma unroll
    for (int c2 = 0; c2 < 2; ++c2)
      #pragma unroll
      for (int i = 0; i < 8; ++i) {
        aq[c2][i]     = qr[c2 * 32 + 2 * koff + i];
        aq[c2][8 + i] = qr[c2 * 32 + 2 * koff + 8 + i];
      }
  }

  issue(0, 0);

  v8f acc[4] = {v8f{}, v8f{}, v8f{}, v8f{}};
  float mrow[8], lrow[8];
  #pragma unroll
  for (int r = 0; r < 8; ++r) { mrow[r] = -1e30f; lrow[r] = 0.f; }
  const float scale = 0.125f; // 64^-0.5

  for (int step = 0; step < kS / 32; ++step) {
    const int buf = step & 1;
    wait_async0();
    __syncthreads();
    if (step + 1 < kS / 32) issue((step + 1) * 32, buf ^ 1);

    // GEMM1: scores for 32 new t columns from LDS K tile.
    v8f sc[2] = {v8f{}, v8f{}};
    #pragma unroll
    for (int half = 0; half < 2; ++half) {
      const __bf16* krow = &kbuf[buf][(half * 16 + col) * 64];
      #pragma unroll
      for (int c2 = 0; c2 < 2; ++c2) {
        const v16bf* bp = (const v16bf*)(krow + c2 * 32 + 16 * hl);
        sc[half] = WMMA_BF16(aq[c2], *bp, sc[half]);
      }
    }
    // Online softmax; row r lives across the 16 lanes of this half.
    #pragma unroll
    for (int r = 0; r < 8; ++r) {
      const float a = sc[0][r] * scale;
      const float b = sc[1][r] * scale;
      float mx = fmaxf(a, b);
      #pragma unroll
      for (int m = 1; m < 16; m <<= 1) mx = fmaxf(mx, __shfl_xor(mx, m, 32));
      const float mn = fmaxf(mrow[r], mx);
      const float cf = __expf(mrow[r] - mn);
      const float pa = __expf(a - mn);
      const float pb = __expf(b - mn);
      float ls = pa + pb;
      #pragma unroll
      for (int m = 1; m < 16; m <<= 1) ls += __shfl_xor(ls, m, 32);
      lrow[r] = lrow[r] * cf + ls;
      mrow[r] = mn;
      #pragma unroll
      for (int dt = 0; dt < 4; ++dt) acc[dt][r] *= cf;
      const int mi = r + 8 * hl;
      pbuf[wid][mi][col]      = (__bf16)pa;
      pbuf[wid][mi][16 + col] = (__bf16)pb;
    }
    wait_ds0(); // same-wave DS in order: P stores complete before A-read
    v16bf ap;
    {
      const __bf16* pr = &pbuf[wid][col][0];
      #pragma unroll
      for (int i = 0; i < 8; ++i) {
        ap[i]     = pr[2 * koff + i];
        ap[8 + i] = pr[2 * koff + 8 + i];
      }
    }
    // GEMM2: out += P(16x32) * V(32x64) from LDS V tile.
    #pragma unroll
    for (int dt = 0; dt < 4; ++dt) {
      const v16bf* bp = (const v16bf*)(&vbuf[buf][(dt * 16 + col) * 32 + 16 * hl]);
      acc[dt] = WMMA_BF16(ap, *bp, acc[dt]);
    }
  }

  const int n  = nh >> 3;
  const int hh = nh & 7;
  #pragma unroll
  for (int dt = 0; dt < 4; ++dt) {
    const int d = dt * 16 + col;
    #pragma unroll
    for (int r = 0; r < 8; ++r) {
      const float val = acc[dt][r] / lrow[r];
      const int s = s0 + r + 8 * hl;
      ao[((size_t)n * kS + s) * kC + hh * kHD + d] = (__bf16)val;
    }
  }
}

// ---------------------------------------------------------------------------
// Projection GEMM + bias + residual: out[n][o][s] (f32).
// Same LDS weight staging as qkv_gemm; contiguous 8-float stores per lane.
// ---------------------------------------------------------------------------
__global__ void proj_gemm(const __bf16* __restrict__ ao, const __bf16* __restrict__ Pw,
                          const float* __restrict__ bias, const float* __restrict__ x,
                          float* __restrict__ out) {
  __shared__ __bf16 bstage[2][2048];
  const int wid  = threadIdx.x >> 5;
  const int lane = threadIdx.x & 31;
  const int oStrip = blockIdx.x & 7;
  const int sGroup = blockIdx.x >> 3;       // 0..127
  const int sTileG = sGroup * 8 + wid;
  const int n  = sTileG >> 6;
  const int s0 = (sTileG & 63) * 16;
  const int hl = lane >> 4;
  const int col = lane & 15;
  const int koff = 8 * hl;

  const __bf16* pbase = Pw + (size_t)oStrip * kKC32 * 2048;
  const int sliceOff = wid * 256 + lane * 8;
  async_copy16(&bstage[0][sliceOff], pbase + sliceOff);

  v8f acc[4] = {v8f{}, v8f{}, v8f{}, v8f{}};
  const __bf16* arow = ao + ((size_t)n * kS + s0 + col) * kC;

  for (int kc = 0; kc < kKC32; ++kc) {
    wait_async0();
    __syncthreads();
    if (kc + 1 < kKC32)
      async_copy16(&bstage[(kc + 1) & 1][sliceOff],
                   pbase + (size_t)(kc + 1) * 2048 + sliceOff);
    __builtin_prefetch(arow + (kc + 2) * 32, 0, 1);
    v16bf a;
    const __bf16* ap = arow + kc * 32;
    #pragma unroll
    for (int i = 0; i < 8; ++i) {
      a[i]     = ap[2 * koff + i];
      a[8 + i] = ap[2 * koff + 8 + i];
    }
    const __bf16* pan = &bstage[kc & 1][0];
    #pragma unroll
    for (int j = 0; j < 4; ++j) {
      const v16bf* bp = (const v16bf*)(pan + j * 512 + lane * 16);
      acc[j] = WMMA_BF16(a, *bp, acc[j]);
    }
  }

  #pragma unroll
  for (int j = 0; j < 4; ++j) {
    const int o = oStrip * 64 + j * 16 + col;
    const float bo = bias[o];
    const size_t base = ((size_t)n * kC + o) * kS + s0 + 8 * hl;
    const float* xr = x + base;
    float* orow = out + base;
    #pragma unroll
    for (int r = 0; r < 8; ++r) orow[r] = acc[j][r] + bo + xr[r];
  }
}

// ---------------------------------------------------------------------------
extern "C" void kernel_launch(void* const* d_in, const int* in_sizes, int n_in,
                              void* d_out, int out_size, void* d_ws, size_t ws_size,
                              hipStream_t stream) {
  (void)in_sizes; (void)n_in; (void)out_size; (void)ws_size;
  const float* x      = (const float*)d_in[0];
  const float* norm_w = (const float*)d_in[1];
  const float* norm_b = (const float*)d_in[2];
  const float* qkv_w  = (const float*)d_in[3];
  const float* qkv_b  = (const float*)d_in[4];
  const float* proj_w = (const float*)d_in[5];
  const float* proj_b = (const float*)d_in[6];
  float* out = (float*)d_out;

  char* ws = (char*)d_ws;
  size_t off = 0;
  auto alloc = [&](size_t bytes) -> void* {
    void* p = ws + off;
    off = (off + bytes + 255) & ~(size_t)255;
    return p;
  };
  float*  mean  = (float*)alloc((size_t)kN * kGroups * sizeof(float));
  float*  rstd  = (float*)alloc((size_t)kN * kGroups * sizeof(float));
  __bf16* h     = (__bf16*)alloc((size_t)kN * kS * kC * 2);
  __bf16* pqkv  = (__bf16*)alloc((size_t)3 * kC * kC * 2);
  __bf16* pproj = (__bf16*)alloc((size_t)kC * kC * 2);
  __bf16* q     = (__bf16*)alloc((size_t)kN * kHeads * kS * kHD * 2);
  __bf16* k     = (__bf16*)alloc((size_t)kN * kHeads * kS * kHD * 2);
  __bf16* v     = (__bf16*)alloc((size_t)kN * kHeads * kHD * kS * 2);
  __bf16* ao    = (__bf16*)alloc((size_t)kN * kS * kC * 2);

  gn_stats<<<kN * kGroups, 256, 0, stream>>>(x, mean, rstd);
  gn_apply<<<kN * kC, 256, 0, stream>>>(x, mean, rstd, norm_w, norm_b, h);
  pack_b_panels<<<(3 * kC * kC) / 256, 256, 0, stream>>>(qkv_w, pqkv, 3 * kC, kC);
  pack_b_panels<<<(kC * kC) / 256, 256, 0, stream>>>(proj_w, pproj, kC, kC);
  // blocks = 128 s-groups x 24 o-strips; 8 waves (s-tiles) per block
  qkv_gemm<<<3072, 256, 0, stream>>>(h, pqkv, qkv_b, q, k, v);
  // blocks = 128 (n,h) x 16 s-groups; 4 waves (16 query rows each) per block
  attn_kernel<<<2048, 128, 0, stream>>>(q, k, v, ao);
  // blocks = 128 s-groups x 8 o-strips
  proj_gemm<<<1024, 256, 0, stream>>>(ao, pproj, proj_b, x, out);
}
